// LucidRains_360777253454
// MI455X (gfx1250) — matrix operand order repaired and checked by reading, not compiled, not run on hardware
//
#include <hip/hip_runtime.h>

// ---- problem dims ----
#define SEQ    8192
#define DMODEL 512
#define NHEADS 8
#define NKV    2
#define GQ     4
#define HD     64
#define BALL   64
#define NBLK   128
#define SM_SCALE 0.125f   // 64^-0.5
#define CPAD   160        // NB+1=129 padded to 160 (mult of 32)

typedef __bf16 bf16_t;
typedef __attribute__((ext_vector_type(16))) __bf16 v16bf;
typedef __attribute__((ext_vector_type(8)))  float  v8f;

// GEMM flags
#define GF_RELU   2
#define GF_STBF   4
#define GF_STF32  8

__device__ __forceinline__ bf16_t f2bf(float f) {
  unsigned u = __float_as_uint(f);
  u += 0x7FFFu + ((u >> 16) & 1u);           // round to nearest even
  unsigned short h = (unsigned short)(u >> 16);
  bf16_t r; __builtin_memcpy(&r, &h, 2); return r;
}
__device__ __forceinline__ float bf2f(bf16_t b) {
  unsigned short h; __builtin_memcpy(&h, &b, 2);
  return __uint_as_float(((unsigned)h) << 16);
}

// ------------------------------------------------------------------
// 0. L2 warm via CDNA5 async global->LDS loads (pure prefetch, data discarded)
// ------------------------------------------------------------------
__global__ void k_l2_warm(const float* __restrict__ w, long long nbytes) {
  __shared__ char buf[4096];
  (void)buf;
  unsigned ldsaddr = threadIdx.x * 16u;
  long long stride = (long long)gridDim.x * blockDim.x * 16;
  long long base = ((long long)blockIdx.x * blockDim.x + threadIdx.x) * 16;
  for (long long off = base; off < nbytes; off += stride) {
    unsigned long long ga = (unsigned long long)(const void*)w + (unsigned long long)off;
    asm volatile("global_load_async_to_lds_b128 %0, %1, off"
                 :: "v"(ldsaddr), "v"(ga) : "memory");
  }
  asm volatile("s_wait_asynccnt 0x0" ::: "memory");
}

// ------------------------------------------------------------------
// 1. per-ball mean of pos
// ------------------------------------------------------------------
__global__ void k_ball_mean(const float* __restrict__ pos, float* __restrict__ means) {
  __shared__ float sh[BALL * 3];
  int b = blockIdx.x, t = threadIdx.x;
  sh[t * 3 + 0] = pos[(b * BALL + t) * 3 + 0];
  sh[t * 3 + 1] = pos[(b * BALL + t) * 3 + 1];
  sh[t * 3 + 2] = pos[(b * BALL + t) * 3 + 2];
  __syncthreads();
  if (t < 3) {
    float s = 0.f;
    for (int i = 0; i < BALL; ++i) s += sh[i * 3 + t];
    means[b * 3 + t] = s * (1.f / BALL);
  }
}

// ------------------------------------------------------------------
// 2. x + rel@pe_w + pe_b, RMSNorm -> xn (f32 + bf16)
// ------------------------------------------------------------------
__global__ void k_preproc(const float* __restrict__ x, const float* __restrict__ pos,
                          const float* __restrict__ means, const float* __restrict__ pe_w,
                          const float* __restrict__ pe_b, const float* __restrict__ gamma,
                          float* __restrict__ xn_f, bf16_t* __restrict__ xn_b) {
  __shared__ float red[128];
  int n = blockIdx.x, t = threadIdx.x;
  int b = n >> 6;
  float r0 = pos[n * 3 + 0] - means[b * 3 + 0];
  float r1 = pos[n * 3 + 1] - means[b * 3 + 1];
  float r2 = pos[n * 3 + 2] - means[b * 3 + 2];
  float v[4]; float ss = 0.f;
  int d0 = t * 4;
#pragma unroll
  for (int i = 0; i < 4; ++i) {
    int d = d0 + i;
    float val = x[n * DMODEL + d] + r0 * pe_w[d] + r1 * pe_w[DMODEL + d]
              + r2 * pe_w[2 * DMODEL + d] + pe_b[d];
    v[i] = val; ss += val * val;
  }
  red[t] = ss; __syncthreads();
  for (int o = 64; o; o >>= 1) { if (t < o) red[t] += red[t + o]; __syncthreads(); }
  float rinv = rsqrtf(red[0] / (float)DMODEL + 1e-6f);
#pragma unroll
  for (int i = 0; i < 4; ++i) {
    int d = d0 + i;
    float nv = v[i] * rinv * gamma[d];
    xn_f[n * DMODEL + d] = nv;
    xn_b[n * DMODEL + d] = f2bf(nv);
  }
}

// ------------------------------------------------------------------
// 3. f32 -> bf16 matrix convert with optional column zero-pad
// ------------------------------------------------------------------
__global__ void k_cvt(const float* __restrict__ src, bf16_t* __restrict__ dst,
                      int rows, int cs, int cd) {
  long long i = (long long)blockIdx.x * blockDim.x + threadIdx.x;
  long long tot = (long long)rows * cd;
  if (i >= tot) return;
  int r = (int)(i / cd), c = (int)(i % cd);
  dst[i] = (c < cs) ? f2bf(src[(long long)r * cs + c]) : f2bf(0.f);
}

// ------------------------------------------------------------------
// 4. generic batched bf16 WMMA GEMM: C = A@B (+bias) (relu), stores f32/bf16
//    A: [M,K] rm, B: [K,N] rm, C: [M,N]. Wave computes 64x16 tile:
//    4 v_wmma_f32_16x16x32_bf16 per K-step sharing one B fragment (4x B reuse).
//    Requires M%64==0, N%16==0, K%32==0, lda/ldb multiples of 8.
// ------------------------------------------------------------------
__global__ void k_wmma_gemm(const bf16_t* __restrict__ A, const bf16_t* __restrict__ B,
                            const float* __restrict__ bias,
                            float* __restrict__ C, bf16_t* __restrict__ Cbf,
                            int M_, int N_, int K_,
                            long long sA, long long sB, long long sC,
                            int lda, int ldb, int ldc, int flags) {
  int lane = threadIdx.x;
  int wave = threadIdx.y;
  int ntn  = N_ >> 4;
  int tiles = (M_ >> 6) * ntn;
  int t = blockIdx.x * blockDim.y + wave;
  if (t >= tiles) return;
  int tm = t / ntn, tn = t - tm * ntn;
  long long bi = blockIdx.y;
  int aoct = lane >> 4;                    // lanes 0-15: K-octets {0,2}; 16-31: {1,3}
  const bf16_t* Abase = A + bi * sA + (long long)(tm * 64 + (lane & 15)) * lda;
  const bf16_t* Bbase = B + bi * sB;
  long long rstep = (long long)16 * lda;   // 16 rows

  union U { v16bf v; uint4 u[2]; };
  v8f acc[4];
#pragma unroll
  for (int s = 0; s < 4; ++s) acc[s] = (v8f){0.f,0.f,0.f,0.f,0.f,0.f,0.f,0.f};

  for (int k0 = 0; k0 < K_; k0 += 32) {
    U bfr;
    const bf16_t* Brow = Bbase + (long long)(k0 + aoct * 8) * ldb + tn * 16 + (lane & 15);
#pragma unroll
    for (int e = 0; e < 8; ++e) {
      bfr.v[e]     = Brow[(long long)e * ldb];
      bfr.v[e + 8] = Brow[(long long)(e + 16) * ldb];
    }
    const bf16_t* Ap = Abase + k0 + aoct * 8;
    if (k0 + 32 < K_) __builtin_prefetch(Ap + 32, 0, 1);  // global_prefetch_b8
    U afr[4];
#pragma unroll
    for (int s = 0; s < 4; ++s) {
      afr[s].u[0] = *(const uint4*)(const void*)(Ap);
      afr[s].u[1] = *(const uint4*)(const void*)(Ap + 16);
      Ap += rstep;
    }
#pragma unroll
    for (int s = 0; s < 4; ++s)
      acc[s] = __builtin_amdgcn_wmma_f32_16x16x32_bf16(
          false, afr[s].v, false, bfr.v, (short)0, acc[s], false, false);
  }
  int cn = tn * 16 + (lane & 15);
  float bv = bias ? bias[cn] : 0.f;
#pragma unroll
  for (int s = 0; s < 4; ++s) {
    int mbase = tm * 64 + s * 16 + (lane >> 4) * 8;
#pragma unroll
    for (int r = 0; r < 8; ++r) {
      float v = acc[s][r] + bv;
      if (flags & GF_RELU) v = fmaxf(v, 0.f);
      long long off = bi * sC + (long long)(mbase + r) * ldc + cn;
      if (flags & GF_STF32) C[off]   = v;
      if (flags & GF_STBF)  Cbf[off] = f2bf(v);
    }
  }
}

// ------------------------------------------------------------------
// 5. build compress-MLP inputs: kb/vb = k/v + k/vpos, bf16 [NKV, NBLK, 4096]
// ------------------------------------------------------------------
__global__ void k_build_kvb(const float* __restrict__ kf, const float* __restrict__ vf,
                            const float* __restrict__ kpos, const float* __restrict__ vpos,
                            bf16_t* __restrict__ kb, bf16_t* __restrict__ vb) {
  int i = blockIdx.x * blockDim.x + threadIdx.x;   // NKV*NBLK*4096 = 2^20
  int hk  = i >> 19;
  int rem = i & ((1 << 19) - 1);
  int nb = rem >> 12, e = rem & 4095;
  int m = e >> 6, d = e & 63;
  int src  = (nb * BALL + m) * (NKV * HD) + hk * HD + d;
  int pidx = (hk * BALL + m) * HD + d;
  kb[i] = f2bf(kf[src] + kpos[pidx]);
  vb[i] = f2bf(vf[src] + vpos[pidx]);
}

// ------------------------------------------------------------------
// 6. build ckf (pre-TRANSPOSED: [H, HD, CPAD]) and cvf ([H, CPAD, HD])
// ------------------------------------------------------------------
__global__ void k_build_ckf(const float* __restrict__ ck, const float* __restrict__ cv,
                            const float* __restrict__ mem_ck, const float* __restrict__ mem_cv,
                            bf16_t* __restrict__ ckf_t, bf16_t* __restrict__ cvf) {
  int i = blockIdx.x * blockDim.x + threadIdx.x;   // NHEADS*CPAD*HD
  if (i >= NHEADS * CPAD * HD) return;
  int h = i / (CPAD * HD), rem = i % (CPAD * HD);
  int j = rem / HD, d = rem % HD;
  int hk = h >> 2;
  float kv = 0.f, vv = 0.f;
  if (j == 0)         { kv = mem_ck[hk * HD + d]; vv = mem_cv[hk * HD + d]; }
  else if (j <= NBLK) { int s = (hk * NBLK + (j - 1)) * HD + d; kv = ck[s]; vv = cv[s]; }
  ckf_t[((long long)h * HD + d) * CPAD + j] = f2bf(kv);   // B[k=d][n=j]
  cvf[i] = f2bf(vv);                                      // B[k=j][n=d]
}

// ------------------------------------------------------------------
// 7. rope + head-major re-layouts (dup K/V per query head for batching);
//    roped K also stored pre-transposed per ball: krd_t [H, NBLK, HD, BALL]
// ------------------------------------------------------------------
__global__ void k_rope_split(const float* __restrict__ qf, const float* __restrict__ kf,
                             const float* __restrict__ vf,
                             bf16_t* __restrict__ qh_b, bf16_t* __restrict__ qr_b,
                             float* __restrict__ qr_f,
                             bf16_t* __restrict__ krd_t, float* __restrict__ kr_f,
                             bf16_t* __restrict__ vd_b,  float* __restrict__ v_f) {
  int i = blockIdx.x * blockDim.x + threadIdx.x;   // NHEADS*SEQ*32
  int h   = i / (SEQ * 32);
  int rem = i % (SEQ * 32);
  int n = rem >> 5, p = rem & 31;
  int hk = h >> 2;
  float q0 = qf[n * DMODEL + h * HD + 2 * p];
  float q1 = qf[n * DMODEL + h * HD + 2 * p + 1];
  float k0 = kf[n * (NKV * HD) + hk * HD + 2 * p];
  float k1 = kf[n * (NKV * HD) + hk * HD + 2 * p + 1];
  float v0 = vf[n * (NKV * HD) + hk * HD + 2 * p];
  float v1 = vf[n * (NKV * HD) + hk * HD + 2 * p + 1];
  float inv = __powf(10000.f, -(float)(2 * p) / (float)HD);
  float s, c; __sincosf((float)n * inv, &s, &c);
  float qr0 = q0 * c - q1 * s, qr1 = q1 * c + q0 * s;
  float kr0 = k0 * c - k1 * s, kr1 = k1 * c + k0 * s;
  long long base = ((long long)h * SEQ + n) * HD + 2 * p;
  qh_b[base]  = f2bf(q0);  qh_b[base + 1]  = f2bf(q1);
  qr_b[base]  = f2bf(qr0); qr_b[base + 1]  = f2bf(qr1);
  qr_f[base]  = qr0;       qr_f[base + 1]  = qr1;
  vd_b[base]  = f2bf(v0);  vd_b[base + 1]  = f2bf(v1);
  int b = n >> 6, m = n & 63;
  long long tb = (((long long)h * NBLK + b) * HD + 2 * p) * BALL + m;
  krd_t[tb]        = f2bf(kr0);
  krd_t[tb + BALL] = f2bf(kr1);
  if ((h & 3) == 0) {
    long long bk = ((long long)hk * SEQ + n) * HD + 2 * p;
    kr_f[bk] = kr0; kr_f[bk + 1] = kr1;
    v_f[bk]  = v0;  v_f[bk + 1]  = v1;
  }
}

// ------------------------------------------------------------------
// 8. compressed softmax with block-causal mask -> bf16 probs (wave per row)
// ------------------------------------------------------------------
__global__ void k_softmax_c(const float* __restrict__ csim, bf16_t* __restrict__ cprob) {
  int lane = threadIdx.x, wy = threadIdx.y;
  long long row = (long long)blockIdx.x * blockDim.y + wy;  // NHEADS*SEQ rows
  int n = (int)(row % SEQ);
  const float* src = csim + row * CPAD;
  float vals[5]; bool vis[5];
  float mx = -3.4e38f;
#pragma unroll
  for (int u = 0; u < 5; ++u) {
    int j = lane + 32 * u;
    bool vb = (j == 0) || (j >= 1 && j <= NBLK && n >= j * BALL - 1);
    float x = vb ? src[j] * SM_SCALE : -3.4e38f;
    vals[u] = x; vis[u] = vb;
    mx = fmaxf(mx, x);
  }
  for (int o = 16; o; o >>= 1) mx = fmaxf(mx, __shfl_xor(mx, o));
  float sum = 0.f;
#pragma unroll
  for (int u = 0; u < 5; ++u) {
    float e = vis[u] ? __expf(vals[u] - mx) : 0.f;
    vals[u] = e; sum += e;
  }
  for (int o = 16; o; o >>= 1) sum += __shfl_xor(sum, o);
  float rs = 1.f / sum;
  bf16_t* dst = cprob + row * CPAD;
#pragma unroll
  for (int u = 0; u < 5; ++u) dst[lane + 32 * u] = f2bf(vals[u] * rs);
}

// ------------------------------------------------------------------
// 9. top-2 block selection (own block forced), matches lax.top_k tie rule
// ------------------------------------------------------------------
__global__ void k_select(const bf16_t* __restrict__ cprob, int* __restrict__ sel) {
  int i = blockIdx.x * blockDim.x + threadIdx.x;   // NKV*SEQ
  if (i >= NKV * SEQ) return;
  int hk = i / SEQ, n = i % SEQ;
  int own = n >> 6;
  float v1 = -1.f, v2 = -1.f; int i1 = -1, i2 = -1;
  for (int nb = 0; nb < NBLK; ++nb) {
    float s = 0.f;
    for (int g = 0; g < GQ; ++g)
      s += bf2f(cprob[(((long long)(hk * GQ + g)) * SEQ + n) * CPAD + 1 + nb]);
    if (nb == own) s = 1e9f;
    if (s > v1) { v2 = v1; i2 = i1; v1 = s; i1 = nb; }
    else if (s > v2) { v2 = s; i2 = nb; }
  }
  sel[i * 2] = i1; sel[i * 2 + 1] = i2;
}

// ------------------------------------------------------------------
// 10. fine branch: gathered 2-block causal attention, wave per (h,n)
// ------------------------------------------------------------------
__global__ void k_fine(const float* __restrict__ qr_f, const float* __restrict__ kr_f,
                       const float* __restrict__ v_f, const int* __restrict__ sel,
                       float* __restrict__ fout) {
  __shared__ float sq[8][HD];
  __shared__ float sp[8][2 * BALL];
  int lane = threadIdx.x, wy = threadIdx.y;
  int row = blockIdx.x * 8 + wy;            // NHEADS*SEQ rows, grid exact
  int h = row / SEQ, n = row % SEQ, hk = h >> 2;
  long long qb = ((long long)h * SEQ + n) * HD;
  sq[wy][lane]      = qr_f[qb + lane];
  sq[wy][lane + 32] = qr_f[qb + lane + 32];
  __syncthreads();
  int s0 = sel[(hk * SEQ + n) * 2], s1 = sel[(hk * SEQ + n) * 2 + 1];
  float sc[4];
#pragma unroll
  for (int u = 0; u < 4; ++u) {
    int s = lane + 32 * u;
    int blk = (s >> 6) ? s1 : s0;
    int kidx = blk * BALL + (s & 63);
    const float* kr = kr_f + ((long long)hk * SEQ + kidx) * HD;
    float acc = 0.f;
    for (int d = 0; d < HD; d += 4) {
      float4 kk = *(const float4*)(const void*)(kr + d);
      acc += kk.x * sq[wy][d] + kk.y * sq[wy][d + 1] + kk.z * sq[wy][d + 2] + kk.w * sq[wy][d + 3];
    }
    sc[u] = (kidx <= n) ? acc * SM_SCALE : -3.4e38f;
  }
  float mx = fmaxf(fmaxf(sc[0], sc[1]), fmaxf(sc[2], sc[3]));
  for (int o = 16; o; o >>= 1) mx = fmaxf(mx, __shfl_xor(mx, o));
  float sum = 0.f;
#pragma unroll
  for (int u = 0; u < 4; ++u) {
    sc[u] = (sc[u] > -1e38f) ? __expf(sc[u] - mx) : 0.f;
    sum += sc[u];
  }
  for (int o = 16; o; o >>= 1) sum += __shfl_xor(sum, o);
  float rs = 1.f / sum;
#pragma unroll
  for (int u = 0; u < 4; ++u) sp[wy][lane + 32 * u] = sc[u] * rs;
  __syncthreads();
  int d0 = lane * 2;
  float a0 = 0.f, a1 = 0.f;
  for (int s = 0; s < 2 * BALL; ++s) {
    int blk = (s >> 6) ? s1 : s0;
    int kidx = blk * BALL + (s & 63);
    float p = sp[wy][s];
    const float* vr = v_f + ((long long)hk * SEQ + kidx) * HD + d0;
    a0 += p * vr[0]; a1 += p * vr[1];
  }
  fout[qb + d0] = a0; fout[qb + d0 + 1] = a1;
}

// ------------------------------------------------------------------
// 11. sliding (per-ball, non-causal) softmax -> bf16, wave per row of 64
// ------------------------------------------------------------------
__global__ void k_softmax_s(const float* __restrict__ ssim, bf16_t* __restrict__ sattn) {
  int lane = threadIdx.x, wy = threadIdx.y;
  long long row = (long long)blockIdx.x * 8 + wy;   // NHEADS*NBLK*64 rows
  const float* src = ssim + row * 64;
  float a = src[lane] * SM_SCALE, b = src[lane + 32] * SM_SCALE;
  float mx = fmaxf(a, b);
  for (int o = 16; o; o >>= 1) mx = fmaxf(mx, __shfl_xor(mx, o));
  float ea = __expf(a - mx), eb = __expf(b - mx);
  float sum = ea + eb;
  for (int o = 16; o; o >>= 1) sum += __shfl_xor(sum, o);
  float rs = 1.f / sum;
  bf16_t* dst = sattn + row * 64;
  dst[lane] = f2bf(ea * rs); dst[lane + 32] = f2bf(eb * rs);
}

// ------------------------------------------------------------------
// 12. sigmoid-gated combine of the 3 branches -> bf16 [SEQ, 512]
// ------------------------------------------------------------------
__global__ void k_combine(const float* __restrict__ gpre, const float* __restrict__ cout_,
                          const float* __restrict__ fout, const float* __restrict__ sout,
                          bf16_t* __restrict__ comb) {
  int i = blockIdx.x * blockDim.x + threadIdx.x;   // SEQ*DMODEL
  int n = i >> 9, c = i & 511;
  int h = c >> 6, d = c & 63;
  float g0 = 1.f / (1.f + __expf(-gpre[n * 32 + h]));
  float g1 = 1.f / (1.f + __expf(-gpre[n * 32 + 8 + h]));
  float g2 = 1.f / (1.f + __expf(-gpre[n * 32 + 16 + h]));
  long long hb = ((long long)h * SEQ + n) * HD + d;
  comb[i] = f2bf(g0 * cout_[hb] + g1 * fout[hb] + g2 * sout[hb]);
}

// ==================================================================
// host launcher
// ==================================================================
extern "C" void kernel_launch(void* const* d_in, const int* in_sizes, int n_in,
                              void* d_out, int out_size, void* d_ws, size_t ws_size,
                              hipStream_t stream) {
  const float* x      = (const float*)d_in[0];
  const float* pos    = (const float*)d_in[1];
  const float* pe_w   = (const float*)d_in[2];
  const float* pe_b   = (const float*)d_in[3];
  const float* gamma  = (const float*)d_in[4];
  const float* wq     = (const float*)d_in[5];
  const float* wk     = (const float*)d_in[6];
  const float* wv     = (const float*)d_in[7];
  const float* kpos   = (const float*)d_in[8];
  const float* vpos   = (const float*)d_in[9];
  const float* kc_w1  = (const float*)d_in[10];
  const float* kc_b1  = (const float*)d_in[11];
  const float* kc_w2  = (const float*)d_in[12];
  const float* kc_b2  = (const float*)d_in[13];
  const float* vc_w1  = (const float*)d_in[14];
  const float* vc_b1  = (const float*)d_in[15];
  const float* vc_w2  = (const float*)d_in[16];
  const float* vc_b2  = (const float*)d_in[17];
  const float* mem_ck = (const float*)d_in[18];
  const float* mem_cv = (const float*)d_in[19];
  const float* wg     = (const float*)d_in[20];
  const float* wo     = (const float*)d_in[21];

  char* wsb = (char*)d_ws;
  size_t off = 0;
  auto alloc = [&](size_t bytes) -> void* {
    off = (off + 255) & ~(size_t)255;
    void* p = wsb + off; off += bytes; return p;
  };

  float*  means  = (float*) alloc((size_t)NBLK * 3 * 4);
  float*  xn_f   = (float*) alloc((size_t)SEQ * DMODEL * 4);
  bf16_t* xn_b   = (bf16_t*)alloc((size_t)SEQ * DMODEL * 2);
  bf16_t* wq_b   = (bf16_t*)alloc((size_t)DMODEL * DMODEL * 2);
  bf16_t* wk_b   = (bf16_t*)alloc((size_t)DMODEL * 128 * 2);
  bf16_t* wv_b   = (bf16_t*)alloc((size_t)DMODEL * 128 * 2);
  bf16_t* wg_b   = (bf16_t*)alloc((size_t)DMODEL * 32 * 2);
  bf16_t* w1k_b  = (bf16_t*)alloc((size_t)4096 * 4096 * 2);
  bf16_t* w2k_b  = (bf16_t*)alloc((size_t)4096 * 64 * 2);
  bf16_t* w1v_b  = (bf16_t*)alloc((size_t)4096 * 4096 * 2);
  bf16_t* w2v_b  = (bf16_t*)alloc((size_t)4096 * 64 * 2);
  bf16_t* wo_b   = (bf16_t*)alloc((size_t)DMODEL * DMODEL * 2);
  float*  qf     = (float*) alloc((size_t)SEQ * DMODEL * 4);
  float*  kf     = (float*) alloc((size_t)SEQ * 128 * 4);
  float*  vf     = (float*) alloc((size_t)SEQ * 128 * 4);
  float*  gpre   = (float*) alloc((size_t)SEQ * 32 * 4);
  bf16_t* kb_b   = (bf16_t*)alloc((size_t)NKV * NBLK * 4096 * 2);
  bf16_t* vb_b   = (bf16_t*)alloc((size_t)NKV * NBLK * 4096 * 2);
  bf16_t* h1k_b  = (bf16_t*)alloc((size_t)NKV * NBLK * 4096 * 2);
  bf16_t* h1v_b  = (bf16_t*)alloc((size_t)NKV * NBLK * 4096 * 2);
  float*  ck     = (float*) alloc((size_t)NKV * NBLK * HD * 4);
  float*  cv     = (float*) alloc((size_t)NKV * NBLK * HD * 4);
  bf16_t* ckf_t  = (bf16_t*)alloc((size_t)NHEADS * HD * CPAD * 2);
  bf16_t* cvf_b  = (bf16_t*)alloc((size_t)NHEADS * CPAD * HD * 2);
  bf16_t* qh_b   = (bf16_t*)alloc((size_t)NHEADS * SEQ * HD * 2);
  bf16_t* qr_b   = (bf16_t*)alloc((size_t)NHEADS * SEQ * HD * 2);
  float*  qr_f   = (float*) alloc((size_t)NHEADS * SEQ * HD * 4);
  bf16_t* krd_t  = (bf16_t*)alloc((size_t)NHEADS * NBLK * HD * BALL * 2);
  float*  kr_f   = (float*) alloc((size_t)NKV * SEQ * HD * 4);
  bf16_t* vd_b   = (bf16_t*)alloc((size_t)NHEADS * SEQ * HD * 2);
  float*  v_f    = (float*) alloc((size_t)NKV * SEQ * HD * 4);
  float*  csim   = (float*) alloc((size_t)NHEADS * SEQ * CPAD * 4);
  bf16_t* cprob  = (bf16_t*)alloc((size_t)NHEADS * SEQ * CPAD * 2);
  int*    sel    = (int*)   alloc((size_t)NKV * SEQ * 2 * 4);
  float*  coutb  = (float*) alloc((size_t)NHEADS * SEQ * HD * 4);
  float*  fout   = (float*) alloc((size_t)NHEADS * SEQ * HD * 4);
  float*  sout   = (float*) alloc((size_t)NHEADS * SEQ * HD * 4);
  float*  ssim   = (float*) alloc((size_t)NHEADS * NBLK * 64 * 64 * 4);
  bf16_t* sattn  = (bf16_t*)alloc((size_t)NHEADS * NBLK * 64 * 64 * 2);
  bf16_t* comb   = (bf16_t*)alloc((size_t)SEQ * DMODEL * 2);

  auto cvt = [&](const float* s, bf16_t* d, int rows, int cs, int cd) {
    long long tot = (long long)rows * cd;
    k_cvt<<<(unsigned)((tot + 255) / 256), 256, 0, stream>>>(s, d, rows, cs, cd);
  };
  auto gemm = [&](const bf16_t* A, const bf16_t* B, const float* bias,
                  float* C, bf16_t* Cbf, int M_, int N_, int K_,
                  long long sA, long long sB, long long sC,
                  int lda, int ldb, int ldc, int flags, int batch) {
    int tiles = (M_ / 64) * (N_ / 16);
    dim3 grid((tiles + 7) / 8, batch, 1), block(32, 8, 1);
    k_wmma_gemm<<<grid, block, 0, stream>>>(A, B, bias, C, Cbf, M_, N_, K_,
                                            sA, sB, sC, lda, ldb, ldc, flags);
  };

  // 0: warm L2 with the big MLP weights via CDNA5 async global->LDS loads
  k_l2_warm<<<256, 256, 0, stream>>>(kc_w1, (long long)4096 * 4096 * 4);
  k_l2_warm<<<256, 256, 0, stream>>>(vc_w1, (long long)4096 * 4096 * 4);

  // 1-2: pos means, preproc/RMSNorm
  k_ball_mean<<<NBLK, BALL, 0, stream>>>(pos, means);
  k_preproc<<<SEQ, 128, 0, stream>>>(x, pos, means, pe_w, pe_b, gamma, xn_f, xn_b);

  // 3: weight conversions
  cvt(wq, wq_b, DMODEL, DMODEL, DMODEL);
  cvt(wk, wk_b, DMODEL, 128, 128);
  cvt(wv, wv_b, DMODEL, 128, 128);
  cvt(wg, wg_b, DMODEL, 24, 32);
  cvt(kc_w1, w1k_b, 4096, 4096, 4096);
  cvt(kc_w2, w2k_b, 4096, 64, 64);
  cvt(vc_w1, w1v_b, 4096, 4096, 4096);
  cvt(vc_w2, w2v_b, 4096, 64, 64);
  cvt(wo, wo_b, DMODEL, DMODEL, DMODEL);

  // 4: q/k/v/gate projections (WMMA)
  gemm(xn_b, wq_b, nullptr, qf, nullptr, SEQ, DMODEL, DMODEL, 0, 0, 0, DMODEL, DMODEL, DMODEL, GF_STF32, 1);
  gemm(xn_b, wk_b, nullptr, kf, nullptr, SEQ, 128, DMODEL, 0, 0, 0, DMODEL, 128, 128, GF_STF32, 1);
  gemm(xn_b, wv_b, nullptr, vf, nullptr, SEQ, 128, DMODEL, 0, 0, 0, DMODEL, 128, 128, GF_STF32, 1);
  gemm(xn_b, wg_b, nullptr, gpre, nullptr, SEQ, 32, DMODEL, 0, 0, 0, DMODEL, 32, 32, GF_STF32, 1);

  // 5-6: compress MLP (batched over kv heads, shared weights), ckf/cvf build
  k_build_kvb<<<(NKV * NBLK * 4096) / 256, 256, 0, stream>>>(kf, vf, kpos, vpos, kb_b, vb_b);
  gemm(kb_b, w1k_b, kc_b1, nullptr, h1k_b, NBLK, 4096, 4096, (long long)NBLK * 4096, 0,
       (long long)NBLK * 4096, 4096, 4096, 4096, GF_RELU | GF_STBF, NKV);
  gemm(h1k_b, w2k_b, kc_b2, ck, nullptr, NBLK, 64, 4096, (long long)NBLK * 4096, 0,
       (long long)NBLK * 64, 4096, 64, 64, GF_STF32, NKV);
  gemm(vb_b, w1v_b, vc_b1, nullptr, h1v_b, NBLK, 4096, 4096, (long long)NBLK * 4096, 0,
       (long long)NBLK * 4096, 4096, 4096, 4096, GF_RELU | GF_STBF, NKV);
  gemm(h1v_b, w2v_b, vc_b2, cv, nullptr, NBLK, 64, 4096, (long long)NBLK * 4096, 0,
       (long long)NBLK * 64, 4096, 64, 64, GF_STF32, NKV);
  k_build_ckf<<<(NHEADS * CPAD * HD + 255) / 256, 256, 0, stream>>>(ck, cv, mem_ck, mem_cv, ckf_t, cvf_b);

  // 7: rope + layouts
  k_rope_split<<<(NHEADS * SEQ * 32) / 256, 256, 0, stream>>>(qf, kf, vf, qh_b, qr_b, qr_f,
                                                              krd_t, kr_f, vd_b, v_f);

  // 8-9: compressed attention (Q·CK^T WMMA, softmax, P·CV WMMA) + selection
  gemm(qh_b, ckf_t, nullptr, csim, nullptr, SEQ, CPAD, HD, (long long)SEQ * HD,
       (long long)HD * CPAD, (long long)SEQ * CPAD, HD, CPAD, CPAD, GF_STF32, NHEADS);
  { dim3 b(32, 8, 1); k_softmax_c<<<(NHEADS * SEQ) / 8, b, 0, stream>>>(csim, cprob); }
  k_select<<<(NKV * SEQ + 255) / 256, 256, 0, stream>>>(cprob, sel);
  gemm(cprob, cvf_b, nullptr, coutb, nullptr, SEQ, HD, CPAD, (long long)SEQ * CPAD,
       (long long)CPAD * HD, (long long)SEQ * HD, CPAD, HD, HD, GF_STF32, NHEADS);

  // 10: fine branch
  { dim3 b(32, 8, 1); k_fine<<<(NHEADS * SEQ) / 8, b, 0, stream>>>(qr_f, kr_f, v_f, sel, fout); }

  // 11: sliding branch (batched 64x64x64 WMMA GEMMs per ball, K pre-transposed)
  gemm(qr_b, krd_t, nullptr, ssim, nullptr, 64, 64, 64, 4096, 4096, 4096,
       64, 64, 64, GF_STF32, NHEADS * NBLK);
  { dim3 b(32, 8, 1); k_softmax_s<<<(NHEADS * NBLK * 64) / 8, b, 0, stream>>>(ssim, sattn); }
  gemm(sattn, vd_b, nullptr, sout, nullptr, 64, 64, 64, 4096, 4096, 4096,
       64, 64, 64, GF_STF32, NHEADS * NBLK);

  // 12: gated combine + output projection
  k_combine<<<(SEQ * DMODEL) / 256, 256, 0, stream>>>(gpre, coutb, fout, sout, comb);
  gemm(comb, wo_b, nullptr, (float*)d_out, nullptr, SEQ, DMODEL, DMODEL, 0, 0, 0,
       DMODEL, DMODEL, DMODEL, GF_STF32, 1);
  (void)in_sizes; (void)n_in; (void)out_size; (void)ws_size;
}
